// ClusterLoss_local_5102421147887
// MI455X (gfx1250) — matrix operand order repaired and checked by reading, not compiled, not run on hardware
//
#include <hip/hip_runtime.h>
#include <hip/hip_bf16.h>
#include <math.h>

// Problem constants (from reference): P=64, K=8, H=32, D=2048
#define PP   64
#define KK   8
#define HH   32
#define DD   2048
#define CROWS (PP * HH)          // 2048 center stripe rows
#define FROWS (PP * KK * HH)     // 16384 feature stripe rows

typedef __attribute__((ext_vector_type(16))) __bf16 v16bf;
typedef __attribute__((ext_vector_type(8)))  __bf16 v8bf;
typedef __attribute__((ext_vector_type(8)))  float  v8f;

// ---------------------------------------------------------------------------
// WMMA helpers (CDNA5 v_wmma_f32_16x16x32_bf16, wave32)
// ---------------------------------------------------------------------------
__device__ __forceinline__ v8f wmma_bf16(v16bf a, v16bf b, v8f c) {
  // (neg_a, A, neg_b, B, c_mod, C, reuse_a, reuse_b)
  return __builtin_amdgcn_wmma_f32_16x16x32_bf16(false, a, false, b,
                                                 (short)0, c, false, false);
}

// A-matrix 16x32 bf16 fragment from row-major [rows, ld]:
// lane&15 -> M row; lanes<16 hold K[k..k+15]; lanes>=16 hold K[k+16..k+31].
__device__ __forceinline__ v16bf load_a_frag(const __bf16* base, int rowBase,
                                             int k, int lane, int ld) {
  const __bf16* p = base + (size_t)(rowBase + (lane & 15)) * ld;
  int h = (lane >> 4) * 16;
  v8bf lo = *(const v8bf*)(p + k + h);
  v8bf hi = *(const v8bf*)(p + k + h + 8);
  return __builtin_shufflevector(lo, hi, 0, 1, 2, 3, 4, 5, 6, 7,
                                 8, 9, 10, 11, 12, 13, 14, 15);
}

// B-matrix 32x16 bf16 fragment; B(k,n) = X[n][k] for row-major X (=> A*X^T).
// lane&15 -> N col; lanes<16 hold K[k..k+15]; lanes>=16 hold K[k+16..k+31].
__device__ __forceinline__ v16bf load_b_frag(const __bf16* base, int colBase,
                                             int k, int lane, int ld) {
  const __bf16* p = base + (size_t)(colBase + (lane & 15)) * ld
                  + k + (lane >> 4) * 16;
  v8bf lo = *(const v8bf*)(p);
  v8bf hi = *(const v8bf*)(p + 8);
  return __builtin_shufflevector(lo, hi, 0, 1, 2, 3, 4, 5, 6, 7,
                                 8, 9, 10, 11, 12, 13, 14, 15);
}

// Same as load_b_frag but converting fp32 source -> bf16 on the fly.
__device__ __forceinline__ v16bf load_b_frag_f32(const float* base, int colBase,
                                                 int k, int lane, int ld) {
  const float* p = base + (size_t)(colBase + (lane & 15)) * ld
                 + k + (lane >> 4) * 16;
  v16bf r;
#pragma unroll
  for (int i = 0; i < 16; ++i) r[i] = (__bf16)p[i];
  return r;
}

__device__ __forceinline__ float dist_tanh(float nx, float ny, float g) {
  float d2 = nx + ny - 2.0f * g;
  float dd = sqrtf(fmaxf(d2, 1e-12f));
  return tanhf(0.5f * dd);
}

// ---------------------------------------------------------------------------
// K1: centers (mean over K), bf16 copy, fp32 squared row norms
// one block per center stripe row (2048 blocks x 256 threads)
// ---------------------------------------------------------------------------
__global__ void __launch_bounds__(256)
centers_kernel(const float* __restrict__ F, __bf16* __restrict__ C16,
               float* __restrict__ cnorm) {
  __shared__ float red[256];
  int row = blockIdx.x;            // p*32 + h
  int p = row >> 5, h = row & 31;
  int tid = threadIdx.x;
  int col = tid * 8;

  float acc[8];
#pragma unroll
  for (int i = 0; i < 8; ++i) acc[i] = 0.0f;
#pragma unroll
  for (int k = 0; k < KK; ++k) {
    const float* src =
        F + ((size_t)((p * KK + k) * HH + h)) * DD + col;
    float4 a = ((const float4*)src)[0];
    float4 b = ((const float4*)src)[1];
    acc[0] += a.x; acc[1] += a.y; acc[2] += a.z; acc[3] += a.w;
    acc[4] += b.x; acc[5] += b.y; acc[6] += b.z; acc[7] += b.w;
  }
  float sq = 0.0f;
  __bf16* dst = C16 + (size_t)row * DD + col;
#pragma unroll
  for (int i = 0; i < 8; ++i) {
    float m = acc[i] * 0.125f;
    dst[i] = (__bf16)m;
    sq += m * m;
  }
  red[tid] = sq;
  __syncthreads();
  for (int off = 128; off > 0; off >>= 1) {
    if (tid < off) red[tid] += red[tid + off];
    __syncthreads();
  }
  if (tid == 0) cnorm[row] = red[0];
}

// ---------------------------------------------------------------------------
// K1b: squared norms of all feature stripe rows (16384 blocks x 256 threads)
// ---------------------------------------------------------------------------
__global__ void __launch_bounds__(256)
fnorm_kernel(const float* __restrict__ F, float* __restrict__ fnorm) {
  __shared__ float red[256];
  int row = blockIdx.x;
  int tid = threadIdx.x;
  const float* src = F + (size_t)row * DD + tid * 8;
  float4 a = ((const float4*)src)[0];
  float4 b = ((const float4*)src)[1];
  float s = a.x * a.x + a.y * a.y + a.z * a.z + a.w * a.w +
            b.x * b.x + b.y * b.y + b.z * b.z + b.w * b.w;
  red[tid] = s;
  __syncthreads();
  for (int off = 128; off > 0; off >>= 1) {
    if (tid < off) red[tid] += red[tid + off];
    __syncthreads();
  }
  if (tid == 0) fnorm[row] = red[0];
}

// ---------------------------------------------------------------------------
// K2: inter Gram (centers x centers^T, 2048^3) with WMMA bf16.
// 1024 waves; each wave computes a 64x64 block (4x4 WMMA tiles), K-loop 64.
// Output written tile-major: [Mi(64), Nj(64), mi(32), nj(32)] after tanh.
// ---------------------------------------------------------------------------
__global__ void __launch_bounds__(256)
inter_gemm_kernel(const __bf16* __restrict__ C16,
                  const float* __restrict__ cnorm,
                  float* __restrict__ dtile) {
  int lane = threadIdx.x & 31;
  int wave = blockIdx.x * 8 + (threadIdx.x >> 5);  // 0..1023
  int tr = (wave >> 5) * 64;
  int tc = (wave & 31) * 64;

  v8f acc[4][4] = {};
  for (int k = 0; k < DD; k += 32) {
    v16bf a[4], b[4];
#pragma unroll
    for (int r = 0; r < 4; ++r) a[r] = load_a_frag(C16, tr + r * 16, k, lane, DD);
#pragma unroll
    for (int c = 0; c < 4; ++c) b[c] = load_b_frag(C16, tc + c * 16, k, lane, DD);
#pragma unroll
    for (int r = 0; r < 4; ++r)
#pragma unroll
      for (int c = 0; c < 4; ++c)
        acc[r][c] = wmma_bf16(a[r], b[c], acc[r][c]);
  }

  int half = lane >> 4;
  int nloc = lane & 15;
#pragma unroll
  for (int r = 0; r < 4; ++r) {
#pragma unroll
    for (int c = 0; c < 4; ++c) {
      int j = tc + c * 16 + nloc;
      float nj = cnorm[j];
      int Nj = j >> 5, njl = j & 31;
#pragma unroll
      for (int e = 0; e < 8; ++e) {
        int i = tr + r * 16 + half * 8 + e;
        float t = dist_tanh(cnorm[i], nj, acc[r][c][e]);
        int Mi = i >> 5, mi = i & 31;
        dtile[(((size_t)(Mi * PP + Nj)) << 10) + (mi << 5) + njl] = t;
      }
    }
  }
}

// ---------------------------------------------------------------------------
// K3: intra Gram per class (centers[p] 32x2048 x feats[p] 256x2048^T).
// 512 waves; wave = (p, colBlock), 2x2 WMMA tiles, B converted fp32->bf16.
// Output tile-major: [p(64), kimg(8), mi(32), nj(32)] after tanh.
// ---------------------------------------------------------------------------
__global__ void __launch_bounds__(256)
intra_gemm_kernel(const __bf16* __restrict__ C16, const float* __restrict__ F,
                  const float* __restrict__ cnorm,
                  const float* __restrict__ fnorm,
                  float* __restrict__ dtile) {
  int lane = threadIdx.x & 31;
  int wave = blockIdx.x * 8 + (threadIdx.x >> 5);  // 0..511
  int p = wave >> 3;
  int cb = wave & 7;
  int aRow = p * HH;                 // center stripe rows of class p
  int bRow = p * (KK * HH) + cb * 32;  // feature stripe rows

  v8f acc[2][2] = {};
  for (int k = 0; k < DD; k += 32) {
    v16bf a[2], b[2];
#pragma unroll
    for (int r = 0; r < 2; ++r) a[r] = load_a_frag(C16, aRow + r * 16, k, lane, DD);
#pragma unroll
    for (int c = 0; c < 2; ++c) b[c] = load_b_frag_f32(F, bRow + c * 16, k, lane, DD);
#pragma unroll
    for (int r = 0; r < 2; ++r)
#pragma unroll
      for (int c = 0; c < 2; ++c)
        acc[r][c] = wmma_bf16(a[r], b[c], acc[r][c]);
  }

  int half = lane >> 4;
  int nloc = lane & 15;
#pragma unroll
  for (int r = 0; r < 2; ++r) {
#pragma unroll
    for (int c = 0; c < 2; ++c) {
      int nfull = cb * 32 + c * 16 + nloc;       // 0..255 within class
      float nj = fnorm[p * (KK * HH) + nfull];
      int kimg = nfull >> 5, njl = nfull & 31;
#pragma unroll
      for (int e = 0; e < 8; ++e) {
        int iloc = r * 16 + half * 8 + e;        // 0..31 center stripe
        float t = dist_tanh(cnorm[aRow + iloc], nj, acc[r][c][e]);
        dtile[(((size_t)(p * KK + kimg)) << 10) + (iloc << 5) + njl] = t;
      }
    }
  }
}

// ---------------------------------------------------------------------------
// K4: shortest-path DP over each 32x32 tile. One wave per tile; lane j owns
// column j; 63 anti-diagonal wavefront steps using __shfl_up. Tile staged in
// LDS so the per-lane row index may be dynamic.
// ---------------------------------------------------------------------------
__global__ void __launch_bounds__(256)
dp_kernel(const float* __restrict__ tiles, float* __restrict__ out, int npairs) {
  __shared__ float sm[8 * 1024];
  int warp = threadIdx.x >> 5;
  int lane = threadIdx.x & 31;
  int w = blockIdx.x * 8 + warp;
  if (w >= npairs) return;  // uniform per wave

  const float* t = tiles + ((size_t)w << 10);
  float* s = sm + warp * 1024;
#pragma unroll
  for (int i = 0; i < 32; ++i) s[i * 32 + lane] = t[i * 32 + lane];

  const float INF = __builtin_inff();
  float distv = 0.0f;
  for (int st = 0; st < 63; ++st) {
    float left = __shfl_up(distv, 1);
    int i = st - lane;
    if (i >= 0 && i < 32) {
      float up = (i == 0) ? ((lane == 0) ? 0.0f : INF) : distv;
      float lf = (lane == 0) ? INF : left;
      distv = s[i * 32 + lane] + fminf(up, lf);
    }
  }
  if (lane == 31) out[w] = distv;  // dist at (31, 31)
}

// ---------------------------------------------------------------------------
// K5: finalize: per-class reductions + margin ReLU mean.
// d_out = [loss, intra_max(64), inter_min(64)]
// ---------------------------------------------------------------------------
__global__ void __launch_bounds__(64)
finalize_kernel(const float* __restrict__ interf, const float* __restrict__ intraf,
                float* __restrict__ out) {
  __shared__ float red[64];
  int p = threadIdx.x;
  const float INF = __builtin_inff();
  float imin = INF;
  for (int j = 0; j < PP; ++j)
    if (j != p) imin = fminf(imin, interf[p * PP + j]);
  float imax = -INF;
  for (int k = 0; k < KK; ++k) imax = fmaxf(imax, intraf[p * KK + k]);
  out[1 + p] = imax;
  out[1 + PP + p] = imin;
  red[p] = fmaxf(imax - imin + 10.0f, 0.0f);
  __syncthreads();
  for (int off = 32; off > 0; off >>= 1) {
    if (p < off) red[p] += red[p + off];
    __syncthreads();
  }
  if (p == 0) out[0] = red[0] * (1.0f / (float)PP);
}

// ---------------------------------------------------------------------------
extern "C" void kernel_launch(void* const* d_in, const int* in_sizes, int n_in,
                              void* d_out, int out_size, void* d_ws, size_t ws_size,
                              hipStream_t stream) {
  (void)in_sizes; (void)n_in; (void)out_size; (void)ws_size;
  const float* F = (const float*)d_in[0];  // [P*K, H, D] fp32
  // d_in[1] = targets: reference assumes ordered contiguous labels, unused.

  char* ws = (char*)d_ws;
  size_t off = 0;
  __bf16* centers_b = (__bf16*)(ws + off); off += (size_t)CROWS * DD * 2;   // 8 MB
  float*  cnorm     = (float*)(ws + off);  off += (size_t)CROWS * 4;
  float*  fnorm     = (float*)(ws + off);  off += (size_t)FROWS * 4;
  float*  dinter    = (float*)(ws + off);  off += (size_t)PP * PP * 1024 * 4;  // 16.8 MB
  float*  dintra    = (float*)(ws + off);  off += (size_t)PP * KK * 1024 * 4;  // 2.1 MB
  float*  interf    = (float*)(ws + off);  off += (size_t)PP * PP * 4;
  float*  intraf    = (float*)(ws + off);  off += (size_t)PP * KK * 4;

  centers_kernel<<<CROWS, 256, 0, stream>>>(F, centers_b, cnorm);
  fnorm_kernel<<<FROWS, 256, 0, stream>>>(F, fnorm);
  inter_gemm_kernel<<<128, 256, 0, stream>>>(centers_b, cnorm, dinter);
  intra_gemm_kernel<<<64, 256, 0, stream>>>(centers_b, F, cnorm, fnorm, dintra);
  dp_kernel<<<(PP * PP) / 8, 256, 0, stream>>>(dinter, interf, PP * PP);
  dp_kernel<<<(PP * KK) / 8, 256, 0, stream>>>(dintra, intraf, PP * KK);
  finalize_kernel<<<1, 64, 0, stream>>>(interf, intraf, (float*)d_out);
}
